// ImageReconstructModel_50903952392813
// MI455X (gfx1250) — matrix operand order repaired and checked by reading, not compiled, not run on hardware
//
#include <hip/hip_runtime.h>
#include <hip/hip_bf16.h>

// ---------------------------------------------------------------------------
// Problem constants (from the reference)
// ---------------------------------------------------------------------------
#define BQ   12
#define CCH  64
#define HH   192
#define WW   192
#define DV   128
#define DIN  192      // DV + C
#define HID  256
#define GG   24       // 2*B garments
#define MT   64       // rows per block tile
#define SPAD 264      // LDS row stride in bf16 elements (256 + 8 pad)

typedef __attribute__((ext_vector_type(16))) __bf16 v16bf;
typedef __attribute__((ext_vector_type(8)))  float  v8f;

union FragU { uint4 u[2]; v16bf v; };
union PackU { __bf16 h[4]; uint2 u; };

static __device__ __forceinline__ __bf16 f2bf(float f) { return (__bf16)f; }

// ---------------------------------------------------------------------------
// Kernel 1: convert weights to bf16, column-major [t][n][k]; pad W3 N: 3->16
// ---------------------------------------------------------------------------
__global__ void prep_weights(const float* __restrict__ W1,
                             const float* __restrict__ W2,
                             const float* __restrict__ W3,
                             __bf16* __restrict__ w1b,
                             __bf16* __restrict__ w2b,
                             __bf16* __restrict__ w3b) {
    const int T1 = 6 * HID * DIN;   // 294912
    const int T2 = 6 * HID * HID;   // 393216
    const int T3 = 6 * 16  * HID;   // 24576
    int i = blockIdx.x * blockDim.x + threadIdx.x;
    if (i < T1) {
        int t = i / (HID * DIN); int r = i % (HID * DIN);
        int n = r / DIN;         int k = r % DIN;
        w1b[i] = f2bf(W1[((long)t * DIN + k) * HID + n]);
    } else if (i < T1 + T2) {
        int e = i - T1;
        int t = e / (HID * HID); int r = e % (HID * HID);
        int n = r / HID;         int k = r % HID;
        w2b[e] = f2bf(W2[((long)t * HID + k) * HID + n]);
    } else if (i < T1 + T2 + T3) {
        int e = i - T1 - T2;
        int t = e / (16 * HID);  int r = e % (16 * HID);
        int n = r / HID;         int k = r % HID;
        w3b[e] = (n < 3) ? f2bf(W3[((long)t * HID + k) * 3 + n]) : f2bf(0.0f);
    }
}

// ---------------------------------------------------------------------------
// Kernel 2: roi_align (1 sample per 1x1 bin, bilinear, clamped coords) + mean
// one block per (garment, channel): 1024 samples reduced by 256 threads
// ---------------------------------------------------------------------------
__global__ void garf_kernel(const float* __restrict__ imgs,
                            const float* __restrict__ pros,
                            const int*   __restrict__ imgbatch,
                            float* __restrict__ garf) {
    int b = blockIdx.x;
    int g = b >> 6;          // garment
    int c = b & 63;          // channel
    int ib = imgbatch[g];
    float px = pros[g * 2 + 0];
    float py = pros[g * 2 + 1];
    const float* base = imgs + (((long)ib * CCH + c) * HH) * WW;

    float sum = 0.0f;
    for (int s = threadIdx.x; s < 1024; s += 256) {
        int i = s >> 5, j = s & 31;
        float fy = py - 16.0f + (float)i + 0.5f;
        float fx = px - 16.0f + (float)j + 0.5f;
        fy = fminf(fmaxf(fy, 0.0f), (float)(HH - 1));
        fx = fminf(fmaxf(fx, 0.0f), (float)(WW - 1));
        int y0 = (int)fy, x0 = (int)fx;           // floor (coords >= 0)
        int y1 = min(y0 + 1, HH - 1), x1 = min(x0 + 1, WW - 1);
        float wy = fy - (float)y0, wx = fx - (float)x0;
        float v00 = base[y0 * WW + x0], v01 = base[y0 * WW + x1];
        float v10 = base[y1 * WW + x0], v11 = base[y1 * WW + x1];
        sum += (v00 * (1.0f - wx) + v01 * wx) * (1.0f - wy)
             + (v10 * (1.0f - wx) + v11 * wx) * wy;
    }
    __shared__ float red[256];
    red[threadIdx.x] = sum;
    __syncthreads();
    for (int st = 128; st > 0; st >>= 1) {
        if (threadIdx.x < st) red[threadIdx.x] += red[threadIdx.x + st];
        __syncthreads();
    }
    if (threadIdx.x == 0) garf[g * 64 + c] = red[0] * (1.0f / 1024.0f);
}

// ---------------------------------------------------------------------------
// Fragment loaders (wave32 WMMA layouts per CDNA5 ISA 7.12.2)
// ---------------------------------------------------------------------------
// A 16x32 bf16: lane 0-15 = rows, khalf=lane>>4; element i in v16bf holds
// K = kbase + (i&8)*2 + khalf*8 + (i&7)  -> two contiguous 16B LDS loads.
static __device__ __forceinline__ v16bf load_afrag(const __bf16* buf, int mi,
                                                   int lane, int kbase) {
    int rloc  = mi * 16 + (lane & 15);
    int khalf = lane >> 4;
    const __bf16* p = buf + rloc * SPAD + kbase + khalf * 8;
    FragU f;
    f.u[0] = *(const uint4*)(p);
    f.u[1] = *(const uint4*)(p + 16);
    return f.v;
}

// B 32x16 bf16 from column-major weights [n][K]: lane&15 = column,
// element i holds K = kbase + (i&8)*2 + (lane>>4)*8 + (i&7)  -> two 16B loads.
static __device__ __forceinline__ v16bf load_bfrag(const __bf16* wcol, int ktot,
                                                   int ncol, int lane, int kbase) {
    const __bf16* p = wcol + (long)ncol * ktot + kbase + (lane >> 4) * 8;
    FragU f;
    f.u[0] = *(const uint4*)(p);
    f.u[1] = *(const uint4*)(p + 16);
    return f.v;
}

// ---------------------------------------------------------------------------
// Kernel 3: per-type expert MLP, 64-row tiles, all GEMMs via bf16 WMMA
// block = 256 threads = 8 waves.
// Wave w owns n-tiles {2w, 2w+1} and ALL four m-tiles, so each B fragment
// (weight tile) is loaded once per block and reused by 4 WMMAs -> 4x less
// L2 weight traffic than one-m-tile-per-wave tiling.
// ---------------------------------------------------------------------------
__global__ void __launch_bounds__(256)
moe_mlp(const float* __restrict__ x_verts,
        const int*   __restrict__ gar_of_vert,
        const float* __restrict__ garf,
        const __bf16* __restrict__ w1b,
        const __bf16* __restrict__ w2b,
        const __bf16* __restrict__ w3b,
        const int* i0, const int* i1, const int* i2,
        const int* i3, const int* i4, const int* i5,
        int n0, int n1, int n2, int n3, int n4, int n5,
        float* __restrict__ out) {
    const int t  = blockIdx.y;
    const int m0 = blockIdx.x;

    const int* idx; int Nt;
    switch (t) {
        case 0:  idx = i0; Nt = n0; break;
        case 1:  idx = i1; Nt = n1; break;
        case 2:  idx = i2; Nt = n2; break;
        case 3:  idx = i3; Nt = n3; break;
        case 4:  idx = i4; Nt = n4; break;
        default: idx = i5; Nt = n5; break;
    }
    if (m0 * MT >= Nt) return;   // uniform early exit (before any barrier)

    extern __shared__ char smem[];
    __bf16* buf0 = (__bf16*)smem;              // 64 x SPAD bf16 (A / L2 out)
    __bf16* buf1 = buf0 + MT * SPAD;           // 64 x SPAD bf16 (L1 out)
    int*    svid = (int*)(buf1 + MT * SPAD);   // 64 vertex ids

    const int tid  = threadIdx.x;
    const int lane = tid & 31;
    const int w    = tid >> 5;

    // ---- Phase A: gather 64 rows of [x_vert(128) | garf(64)] -> bf16 LDS ----
    // 4 threads/row, 48 cols each; 16B-aligned and never straddling the
    // x_vert/garf boundary mid-vector -> pure float4 loads, uint2 bf16 stores.
    {
        int r    = tid >> 2;          // 64 rows
        int part = tid & 3;           // 48 columns each
        int gi   = m0 * MT + r;
        int vid  = -1, g = 0;
        if (gi < Nt) { vid = idx[gi]; g = gar_of_vert[vid]; }
        if (part == 0) svid[r] = vid;
        const float* vsrc = x_verts + (long)(vid < 0 ? 0 : vid) * DV;
        const float* gsrc = garf + g * CCH;
        int c0 = part * 48;
        for (int c = c0; c < c0 + 48; c += 4) {
            float4 v = make_float4(0.f, 0.f, 0.f, 0.f);
            if (vid >= 0)
                v = (c < DV) ? *(const float4*)(vsrc + c)
                             : *(const float4*)(gsrc + (c - DV));
            PackU pk;
            pk.h[0] = f2bf(v.x); pk.h[1] = f2bf(v.y);
            pk.h[2] = f2bf(v.z); pk.h[3] = f2bf(v.w);
            *(uint2*)(buf0 + r * SPAD + c) = pk.u;
        }
    }
    __syncthreads();

    const int njb = w * 2;            // n-tiles owned by this wave

    // ---- Layer 1: H1 = relu(A[64x192] @ W1[192x256]) ----
    {
        const __bf16* wt = w1b + (long)t * HID * DIN;
        v8f acc[2][4];
#pragma unroll
        for (int j = 0; j < 2; ++j)
#pragma unroll
            for (int mi = 0; mi < 4; ++mi) acc[j][mi] = (v8f)(0.0f);

#pragma unroll
        for (int ks = 0; ks < 6; ++ks) {
            v16bf afrag[4];
#pragma unroll
            for (int mi = 0; mi < 4; ++mi)
                afrag[mi] = load_afrag(buf0, mi, lane, ks * 32);
#pragma unroll
            for (int j = 0; j < 2; ++j) {
                int ncol = (njb + j) * 16 + (lane & 15);
                v16bf b = load_bfrag(wt, DIN, ncol, lane, ks * 32);
#pragma unroll
                for (int mi = 0; mi < 4; ++mi)
                    acc[j][mi] = __builtin_amdgcn_wmma_f32_16x16x32_bf16(
                        false, afrag[mi], false, b, (short)0, acc[j][mi], false, false);
            }
        }
        // relu + bf16 + store to buf1
#pragma unroll
        for (int j = 0; j < 2; ++j) {
            int col = (njb + j) * 16 + (lane & 15);
#pragma unroll
            for (int mi = 0; mi < 4; ++mi)
#pragma unroll
                for (int r = 0; r < 8; ++r) {
                    int row = mi * 16 + r + (lane >> 4) * 8;
                    buf1[row * SPAD + col] = f2bf(fmaxf(acc[j][mi][r], 0.0f));
                }
        }
    }
    __syncthreads();

    // ---- Layer 2: H2 = relu(H1[64x256] @ W2[256x256]) ----
    {
        const __bf16* wt = w2b + (long)t * HID * HID;
        v8f acc[2][4];
#pragma unroll
        for (int j = 0; j < 2; ++j)
#pragma unroll
            for (int mi = 0; mi < 4; ++mi) acc[j][mi] = (v8f)(0.0f);

#pragma unroll
        for (int ks = 0; ks < 8; ++ks) {
            v16bf afrag[4];
#pragma unroll
            for (int mi = 0; mi < 4; ++mi)
                afrag[mi] = load_afrag(buf1, mi, lane, ks * 32);
#pragma unroll
            for (int j = 0; j < 2; ++j) {
                int ncol = (njb + j) * 16 + (lane & 15);
                v16bf b = load_bfrag(wt, HID, ncol, lane, ks * 32);
#pragma unroll
                for (int mi = 0; mi < 4; ++mi)
                    acc[j][mi] = __builtin_amdgcn_wmma_f32_16x16x32_bf16(
                        false, afrag[mi], false, b, (short)0, acc[j][mi], false, false);
            }
        }
#pragma unroll
        for (int j = 0; j < 2; ++j) {
            int col = (njb + j) * 16 + (lane & 15);
#pragma unroll
            for (int mi = 0; mi < 4; ++mi)
#pragma unroll
                for (int r = 0; r < 8; ++r) {
                    int row = mi * 16 + r + (lane >> 4) * 8;
                    buf0[row * SPAD + col] = f2bf(fmaxf(acc[j][mi][r], 0.0f));
                }
        }
    }
    __syncthreads();

    // ---- Layer 3: O = H2[64x256] @ W3[256x16pad]; scatter cols 0..2 ----
    if (w < 4) {
        const int mi3 = w;
        const __bf16* wt = w3b + (long)t * 16 * HID;
        v8f acc = (v8f)(0.0f);
        int ncol = lane & 15;
#pragma unroll
        for (int ks = 0; ks < 8; ++ks) {
            v16bf a = load_afrag(buf0, mi3, lane, ks * 32);
            v16bf b = load_bfrag(wt, HID, ncol, lane, ks * 32);
            acc = __builtin_amdgcn_wmma_f32_16x16x32_bf16(
                false, a, false, b, (short)0, acc, false, false);
        }
        int col = lane & 15;
        if (col < 3) {
#pragma unroll
            for (int r = 0; r < 8; ++r) {
                int row = mi3 * 16 + r + (lane >> 4) * 8;
                int vid = svid[row];
                if (vid >= 0) out[(long)vid * 3 + col] = acc[r];
            }
        }
    }
}

// ---------------------------------------------------------------------------
// Host launcher
// ---------------------------------------------------------------------------
extern "C" void kernel_launch(void* const* d_in, const int* in_sizes, int n_in,
                              void* d_out, int out_size, void* d_ws, size_t ws_size,
                              hipStream_t stream) {
    const float* imgs     = (const float*)d_in[0];
    const float* pros     = (const float*)d_in[1];
    const float* x_verts  = (const float*)d_in[2];
    const float* W1       = (const float*)d_in[3];
    const float* W2       = (const float*)d_in[4];
    const float* W3       = (const float*)d_in[5];
    const int*   imgbatch = (const int*)d_in[6];
    const int*   garv     = (const int*)d_in[7];
    const int*   idx[6]   = { (const int*)d_in[8],  (const int*)d_in[9],
                              (const int*)d_in[10], (const int*)d_in[11],
                              (const int*)d_in[12], (const int*)d_in[13] };
    int cnt[6];
    int maxN = 0;
    for (int t2 = 0; t2 < 6; ++t2) {
        cnt[t2] = in_sizes[8 + t2];
        if (cnt[t2] > maxN) maxN = cnt[t2];
    }
    float* outp = (float*)d_out;

    // workspace layout (bytes)
    char* ws = (char*)d_ws;
    float*  garf = (float*)(ws);                           // 24*64*4       = 6144
    __bf16* w1b  = (__bf16*)(ws + 6144);                   // 6*256*192*2   = 589824
    __bf16* w2b  = (__bf16*)(ws + 6144 + 589824);          // 6*256*256*2   = 786432
    __bf16* w3b  = (__bf16*)(ws + 6144 + 589824 + 786432); // 6*16*256*2    = 49152

    // 1) weights -> bf16 column-major
    {
        int total = 6 * HID * DIN + 6 * HID * HID + 6 * 16 * HID;  // 712704
        prep_weights<<<(total + 255) / 256, 256, 0, stream>>>(W1, W2, W3, w1b, w2b, w3b);
    }
    // 2) roi_align + pooled garment features
    garf_kernel<<<GG * CCH, 256, 0, stream>>>(imgs, pros, imgbatch, garf);

    // 3) MoE expert MLP with WMMA
    {
        dim3 grid((maxN + MT - 1) / MT, 6, 1);
        size_t lds = (size_t)(2 * MT * SPAD) * sizeof(__bf16) + MT * sizeof(int);
        moe_mlp<<<grid, 256, lds, stream>>>(x_verts, garv, garf,
                                            w1b, w2b, w3b,
                                            idx[0], idx[1], idx[2], idx[3], idx[4], idx[5],
                                            cnt[0], cnt[1], cnt[2], cnt[3], cnt[4], cnt[5],
                                            outp);
    }
}
